// Head_12652973654564
// MI455X (gfx1250) — matrix operand order repaired and checked by reading, not compiled, not run on hardware
//
#include <hip/hip_runtime.h>
#include <hip/hip_bf16.h>
#include <stdint.h>

// Problem constants (match reference)
#define BB 8
#define TT 2048
#define CC 1024
#define HH 64
#define NKC (CC / 32)                 // 32 K-chunks
#define PERW ((CC / 2) * HH)          // 32768 u32 per packed weight matrix

typedef __attribute__((ext_vector_type(16))) __bf16 v16bf;
typedef __attribute__((ext_vector_type(8)))  float  v8f;

union Frag16 { uint32_t u[8]; v16bf v; };

__device__ __forceinline__ uint32_t pack2bf(float lo, float hi) {
    __hip_bfloat162 h = __float22bfloat162_rn(make_float2(lo, hi));
    uint32_t r;
    __builtin_memcpy(&r, &h, 4);
    return r;
}
__device__ __forceinline__ uint16_t f2bf(float f) {
    uint32_t u = __float_as_uint(f);
    u += 0x7fffu + ((u >> 16) & 1u);          // round-to-nearest-even
    return (uint16_t)(u >> 16);
}
__device__ __forceinline__ v8f wmma_bf16(const Frag16& a, const Frag16& b, v8f c) {
    return __builtin_amdgcn_wmma_f32_16x16x32_bf16(false, a.v, false, b.v,
                                                   (short)0, c, false, false);
}
__device__ __forceinline__ void frag_from_u4(Frag16& f, uint4 lo, uint4 hi) {
    f.u[0] = lo.x; f.u[1] = lo.y; f.u[2] = lo.z; f.u[3] = lo.w;
    f.u[4] = hi.x; f.u[5] = hi.y; f.u[6] = hi.z; f.u[7] = hi.w;
}

// ---------------------------------------------------------------------------
// Kernel 1: fp32 weights -> bf16 B-fragments in fragment-major order:
//   wp[w][chunk=C/32][nc=H/16][lane=32][j=8]  (u32 = packed bf16 pair)
// lane = half*16 + m :  n = nc*16 + m,  k = chunk*32 + (j>>2)*16 + half*8 + 2(j&3)
// ---------------------------------------------------------------------------
__global__ __launch_bounds__(256) void pack_weights_k(
    const float* __restrict__ Wq, const float* __restrict__ Wk,
    const float* __restrict__ Wv, uint32_t* __restrict__ wp)
{
    int idx = blockIdx.x * 256 + threadIdx.x;
    if (idx >= 3 * PERW) return;
    int w    = idx / PERW;
    int rem  = idx - w * PERW;
    int j     = rem & 7;
    int lane  = (rem >> 3) & 31;
    int nc    = (rem >> 8) & 3;
    int chunk = rem >> 10;
    int m = lane & 15, half = lane >> 4;
    int k = chunk * 32 + (j >> 2) * 16 + half * 8 + 2 * (j & 3);
    int n = nc * 16 + m;
    const float* W = (w == 0) ? Wq : (w == 1) ? Wk : Wv;
    wp[idx] = pack2bf(W[(k + 0) * HH + n], W[(k + 1) * HH + n]);
}

// Load one 4-fragment B group (weight w, K-chunk kc) : 8x global_load_b128
__device__ __forceinline__ void load_bgroup(Frag16 (&dst)[4],
                                            const uint32_t* wbase, int w, int kc) {
#pragma unroll
    for (int nc = 0; nc < 4; ++nc) {
        const uint4* pb = (const uint4*)(wbase + (size_t)w * PERW
                                         + (size_t)(kc * 4 + nc) * 256);
        frag_from_u4(dst[nc], pb[0], pb[1]);
    }
}

// One K-chunk: convert A, prefetch next x, 3 groups of 4 WMMAs with the two
// 4-fragment buffers ping-ponged (both live at once -> cannot be coalesced).
// PAR = buffer holding this chunk's group 0 (alternates per chunk).
template <int PAR>
__device__ __forceinline__ void proj_chunk(
    int kc, const float* xrow, const uint32_t* wbase,
    float4 (&xraw)[4], Frag16 (&bb)[2][4], v8f (&acc)[12])
{
    // convert current chunk's A fragment (v_cvt_pk_bf16_f32)
    Frag16 a;
    a.u[0] = pack2bf(xraw[0].x, xraw[0].y);
    a.u[1] = pack2bf(xraw[0].z, xraw[0].w);
    a.u[2] = pack2bf(xraw[1].x, xraw[1].y);
    a.u[3] = pack2bf(xraw[1].z, xraw[1].w);
    a.u[4] = pack2bf(xraw[2].x, xraw[2].y);
    a.u[5] = pack2bf(xraw[2].z, xraw[2].w);
    a.u[6] = pack2bf(xraw[3].x, xraw[3].y);
    a.u[7] = pack2bf(xraw[3].z, xraw[3].w);

    // issue next chunk's x loads early (clamped on last chunk; branch-free)
    int kcn = (kc + 1 < NKC) ? (kc + 1) : (NKC - 1);
    {
        const float4* p = (const float4*)(xrow + kcn * 32);
        xraw[0] = p[0]; xraw[1] = p[1]; xraw[2] = p[4]; xraw[3] = p[5];
    }

    // group 0 (w=0): data preloaded in bb[PAR]; fill other buffer with group 1
    load_bgroup(bb[PAR ^ 1], wbase, 1, kc);
#pragma unroll
    for (int nc = 0; nc < 4; ++nc) acc[nc] = wmma_bf16(a, bb[PAR][nc], acc[nc]);

    // group 1 (w=1): consume bb[PAR^1]; refill bb[PAR] with group 2
    load_bgroup(bb[PAR], wbase, 2, kc);
#pragma unroll
    for (int nc = 0; nc < 4; ++nc) acc[4 + nc] = wmma_bf16(a, bb[PAR ^ 1][nc], acc[4 + nc]);

    // group 2 (w=2): consume bb[PAR]; refill bb[PAR^1] with next chunk's group 0
    load_bgroup(bb[PAR ^ 1], wbase, 0, kcn);
#pragma unroll
    for (int nc = 0; nc < 4; ++nc) acc[8 + nc] = wmma_bf16(a, bb[PAR][nc], acc[8 + nc]);
}

// ---------------------------------------------------------------------------
// Kernel 2: fused q,k,v = x @ {Wq,Wk,Wv}. One wave -> one 16-row tile, all
// three 16x64 outputs; x read once from HBM. Fully software-pipelined.
// ---------------------------------------------------------------------------
__global__ __launch_bounds__(128, 1) void qkv_proj_k(
    const float*   __restrict__ x,  const uint32_t* __restrict__ wp,
    uint16_t* __restrict__ qb, uint16_t* __restrict__ kb, uint32_t* __restrict__ vp)
{
    const int wave = threadIdx.x >> 5;
    const int lane = threadIdx.x & 31;
    const int m    = lane & 15;
    const int half = lane >> 4;
    const int bt0  = blockIdx.x * 64 + wave * 16;

    v8f acc[12];
#pragma unroll
    for (int i = 0; i < 12; ++i) acc[i] = v8f{};

    const float*    xrow  = x + (size_t)(bt0 + m) * CC + half * 8;
    const uint32_t* wbase = wp + lane * 8;

    // prologue: x raw for chunk 0, B group (w=0, kc=0) into buffer 0
    float4 xraw[4];
    {
        const float4* p = (const float4*)(xrow);
        xraw[0] = p[0]; xraw[1] = p[1]; xraw[2] = p[4]; xraw[3] = p[5];
    }
    Frag16 bb[2][4];
    load_bgroup(bb[0], wbase, 0, 0);

    // chunk parity alternates (3 groups per chunk): unroll by 2
    for (int kc2 = 0; kc2 < NKC / 2; ++kc2) {
        proj_chunk<0>(2 * kc2 + 0, xrow, wbase, xraw, bb, acc);
        proj_chunk<1>(2 * kc2 + 1, xrow, wbase, xraw, bb, acc);
    }

    // q and k: row-major bf16 stores (pairs along H stay contiguous)
#pragma unroll
    for (int w = 0; w < 2; ++w) {
        uint16_t* out = (w == 0) ? qb : kb;
#pragma unroll
        for (int nc = 0; nc < 4; ++nc)
#pragma unroll
            for (int r = 0; r < 8; ++r) {
                int t = bt0 + r + 8 * half;                           // C/D layout row
                out[(size_t)t * HH + nc * 16 + m] = f2bf(acc[w * 4 + nc][r]);
            }
    }
    // v: fragment-major store (pairs of adjacent T rows live in one lane)
    {
        int bidx = bt0 / TT, tloc = bt0 % TT;
#pragma unroll
        for (int nc = 0; nc < 4; ++nc)
#pragma unroll
            for (int i = 0; i < 4; ++i) {
                int p  = (tloc >> 1) + half * 4 + i;   // pair index within batch
                int sb = p >> 4, pp = p & 15;          // 32-key block, pair-in-block
                int lane_f = (pp >> 3) * 16 + m;
                int jf     = pp & 7;
                vp[(((size_t)bidx * (TT / 32) + sb) * 4 + nc) * 256 + lane_f * 8 + jf]
                    = pack2bf(acc[8 + nc][2 * i], acc[8 + nc][2 * i + 1]);
            }
    }
}

// ---------------------------------------------------------------------------
// Kernel 3: flash attention. One wave = 16 queries; key blocks of 32.
// All 4 K fragments preloaded before the S-WMMAs (staged waits); all 4 V
// fragments preloaded before the softmax VALU section (latency fully hidden).
// ---------------------------------------------------------------------------
__global__ __launch_bounds__(128, 2) void flash_attn_k(
    const uint32_t* __restrict__ qb, const uint32_t* __restrict__ kb,
    const uint32_t* __restrict__ vp, float* __restrict__ out)
{
    __shared__ uint32_t pLds[4 * 16 * 16];       // 4 waves x (16x32 bf16) = 4 KB
    const int b    = blockIdx.y;
    const int wave = threadIdx.x >> 5;
    const int lane = threadIdx.x & 31;
    const int m    = lane & 15;
    const int half = lane >> 4;
    const int t0   = blockIdx.x * 64 + wave * 16;
    const size_t rowBase = (size_t)b * TT;

    // Q A-fragments: u32 row stride = H/2 = 32; 2x uint4 per chunk
    Frag16 aQ[2];
#pragma unroll
    for (int ch = 0; ch < 2; ++ch) {
        const uint4* pq = (const uint4*)(qb + (rowBase + t0 + m) * 32
                                         + ch * 16 + half * 4);
        frag_from_u4(aQ[ch], pq[0], pq[2]);      // +0 (j0..3) and +8 dwords (j4..7)
    }

    v8f o[4] = {v8f{}, v8f{}, v8f{}, v8f{}};
    float Mr[8], Lr[8];
#pragma unroll
    for (int r = 0; r < 8; ++r) { Mr[r] = -3.0e38f; Lr[r] = 0.0f; }

    uint32_t* myP   = pLds + wave * 256;
    uint16_t* myP16 = (uint16_t*)myP;

    const int nb = (t0 + 15) / 32 + 1;           // causal: key blocks <= query tile
    for (int jb = 0; jb < nb; ++jb) {
        const int s0 = jb * 32;
        if (jb + 1 < nb)                          // prefetch next K block
            __builtin_prefetch(&kb[(rowBase + s0 + 32) * 32], 0, 0);

        // preload all 4 K B-fragments (8x b128 clause), then 4 WMMAs w/ staged waits
        Frag16 bk[4];
#pragma unroll
        for (int i = 0; i < 4; ++i) {
            int sub = i >> 1, ch = i & 1;
            const uint4* pk = (const uint4*)(kb + (rowBase + s0 + sub * 16 + m) * 32
                                             + ch * 16 + half * 4);
            frag_from_u4(bk[i], pk[0], pk[2]);
        }
        v8f s[2] = {v8f{}, v8f{}};
#pragma unroll
        for (int i = 0; i < 4; ++i)
            s[i >> 1] = wmma_bf16(aQ[i & 1], bk[i], s[i >> 1]);

        // preload all 4 V B-fragments now; softmax below hides their latency
        Frag16 bv[4];
#pragma unroll
        for (int nc = 0; nc < 4; ++nc) {
            const uint4* pv = (const uint4*)(vp
                + (((size_t)b * (TT / 32) + jb) * 4 + nc) * 256 + lane * 8);
            frag_from_u4(bv[nc], pv[0], pv[1]);
        }

        // scale + causal mask
        float sv[2][8];
#pragma unroll
        for (int sub = 0; sub < 2; ++sub)
#pragma unroll
            for (int r = 0; r < 8; ++r) {
                float v  = s[sub][r] * 0.125f;               // 1/sqrt(64)
                int key  = s0 + sub * 16 + m;
                int trow = t0 + r + 8 * half;
                sv[sub][r] = (key > trow) ? -1.0e30f : v;
            }

        // online softmax: row max / rescale factor
        float sc[8];
#pragma unroll
        for (int r = 0; r < 8; ++r) {
            float v = fmaxf(sv[0][r], sv[1][r]);
            v = fmaxf(v, __shfl_xor(v, 1));
            v = fmaxf(v, __shfl_xor(v, 2));
            v = fmaxf(v, __shfl_xor(v, 4));
            v = fmaxf(v, __shfl_xor(v, 8));      // stays inside 16-lane half = row set
            float mn = fmaxf(Mr[r], v);
            sc[r] = __expf(Mr[r] - mn);
            Mr[r] = mn;
        }
        // P = exp(S - M): write bf16 to per-wave LDS, accumulate row sums
#pragma unroll
        for (int r = 0; r < 8; ++r) {
            float p0 = __expf(sv[0][r] - Mr[r]);
            float p1 = __expf(sv[1][r] - Mr[r]);
            int row = r + 8 * half;
            myP16[row * 32 + m]      = f2bf(p0);
            myP16[row * 32 + 16 + m] = f2bf(p1);
            float v = p0 + p1;
            v += __shfl_xor(v, 1);
            v += __shfl_xor(v, 2);
            v += __shfl_xor(v, 4);
            v += __shfl_xor(v, 8);
            Lr[r] = Lr[r] * sc[r] + v;
        }
#pragma unroll
        for (int nc = 0; nc < 4; ++nc)
#pragma unroll
            for (int r = 0; r < 8; ++r)
                o[nc][r] *= sc[r];

        // wave-private LDS: cross-lane RAW needs an explicit DS wait (no barrier)
        asm volatile("s_wait_dscnt 0" ::: "memory");

        // read P back as A fragment: consecutive dwords -> 2x ds_load_b128
        Frag16 ap;
        {
            const uint4* pp = (const uint4*)(myP + m * 16 + half * 8);
            frag_from_u4(ap, pp[0], pp[1]);      // +0 and +4 dwords
        }

        // O += P @ V with preloaded fragments
#pragma unroll
        for (int nc = 0; nc < 4; ++nc)
            o[nc] = wmma_bf16(ap, bv[nc], o[nc]);
    }

    // normalize and store fp32 output
#pragma unroll
    for (int nc = 0; nc < 4; ++nc)
#pragma unroll
        for (int r = 0; r < 8; ++r) {
            int t = t0 + r + 8 * half;
            out[(rowBase + t) * HH + nc * 16 + m] = o[nc][r] / Lr[r];
        }
}

// ---------------------------------------------------------------------------
extern "C" void kernel_launch(void* const* d_in, const int* in_sizes, int n_in,
                              void* d_out, int out_size, void* d_ws, size_t ws_size,
                              hipStream_t stream)
{
    const float* x  = (const float*)d_in[0];
    const float* Wq = (const float*)d_in[1];
    const float* Wk = (const float*)d_in[2];
    const float* Wv = (const float*)d_in[3];
    float* out = (float*)d_out;

    // workspace layout
    char* ws = (char*)d_ws;
    uint32_t* wp = (uint32_t*)ws;                                  // 3*32768 u32 = 384 KB
    uint16_t* qb = (uint16_t*)(ws + 393216);                       // 2 MB bf16
    uint16_t* kb = (uint16_t*)(ws + 393216 + 2097152);             // 2 MB bf16
    uint32_t* vp = (uint32_t*)(ws + 393216 + 2 * 2097152);         // 2 MB u32 (frag-major)

    // 1) weights -> bf16 fragment-major
    pack_weights_k<<<(3 * PERW + 255) / 256, 256, 0, stream>>>(Wq, Wk, Wv, wp);

    // 2) fused QKV projection: x read once; 12 WMMAs per K-chunk per wave
    qkv_proj_k<<<dim3((BB * TT) / 64), 128, 0, stream>>>(x, wp, qb, kb, vp);

    // 3) flash attention: 64 queries per block (4 waves x 16) -> 256 blocks
    flash_attn_k<<<dim3(TT / 64, BB), 128, 0, stream>>>(
        (const uint32_t*)qb, (const uint32_t*)kb, vp, out);
}
